// Node_Model_15676630631269
// MI455X (gfx1250) — matrix operand order repaired and checked by reading, not compiled, not run on hardware
//
#include <hip/hip_runtime.h>
#include <hip/hip_bf16.h>

typedef __attribute__((ext_vector_type(16))) __bf16   v16bf;
typedef __attribute__((ext_vector_type(8)))  float    v8f;
typedef __attribute__((ext_vector_type(4)))  unsigned vu4;
typedef __attribute__((ext_vector_type(2)))  float    f2;

#define HID 64
#define W23S 72   /* padded bf16 row stride for W2/W3 staging (144B = 16B-mult) */
#define W4S  88   /* padded N row count for W4 staging (row stride 64 bf16)   */
#define H2S  66   /* padded bf16 row stride for h2 staging (odd word multiplier) */
#define Z1S  68   /* padded f32 row stride for z1 staging */

union BFrag { vu4 q[2]; v16bf v; };
union AFrag { unsigned u[8]; v16bf v; };

__device__ __forceinline__ void atomic_add_f32_asm(float* p, float v) {
    unsigned long long a = (unsigned long long)p;
    asm volatile("global_atomic_add_f32 %0, %1, off" :: "v"(a), "v"(v) : "memory");
}

// ---------------------------------------------------------------------------
// Prep: zero accumulators, convert weights to bf16 with padded layouts.
// ---------------------------------------------------------------------------
__global__ void __launch_bounds__(256) prep_kernel(
    const float* __restrict__ W2, const float* __restrict__ W3,
    const float* __restrict__ W4,
    float* __restrict__ summed, float* __restrict__ cnt,
    __bf16* __restrict__ w2b, __bf16* __restrict__ w3b, __bf16* __restrict__ w4b,
    int N)
{
    long long idx = (long long)blockIdx.x * 256 + threadIdx.x;
    long long tot = (long long)N * HID;
    if (idx < tot) summed[idx] = 0.0f;
    if (idx < N)   cnt[idx] = 0.0f;
    if (idx < 64 * W23S) {
        int k = (int)idx / W23S, n = (int)idx % W23S;
        w2b[idx] = (n < 64) ? (__bf16)W2[n * 64 + k] : (__bf16)0.0f;
        w3b[idx] = (n < 64) ? (__bf16)W3[n * 64 + k] : (__bf16)0.0f;
    }
    if (idx < 64 * W4S) {
        int n = (int)idx / 64, k = (int)idx % 64;
        // rows 2..65 of W4 (the 'agg' block), stored n-major (stride 64), N padded with zeros
        w4b[idx] = (n < 67) ? (__bf16)W4[(2 + k) * 67 + n] : (__bf16)0.0f;
    }
}

// ---------------------------------------------------------------------------
// Edge MLP + scatter-add.  One wave (32 lanes) per tile of 16 edges.
// ---------------------------------------------------------------------------
__global__ void __launch_bounds__(256) edge_kernel(
    const float* __restrict__ x, const int* __restrict__ edge_index,
    const float* __restrict__ edge_attr,
    const float* __restrict__ W1, const float* __restrict__ b1,
    const float* __restrict__ b2, const float* __restrict__ b3,
    const __bf16* __restrict__ w2b, const __bf16* __restrict__ w3b,
    float* __restrict__ summed, float* __restrict__ cnt,
    int E, int numTiles)
{
    __shared__ __bf16 w2s[64 * W23S];
    __shared__ __bf16 w3s[64 * W23S];
    __shared__ float  w1s[3 * 64];
    __shared__ float  b1s[64], b2s[64], b3s[64];
    __shared__ __bf16 h2s[8][16 * H2S];
    __shared__ int    dsts[8][16];

    int t = threadIdx.x;
    for (int i = t; i < 64 * W23S; i += 256) { w2s[i] = w2b[i]; w3s[i] = w3b[i]; }
    if (t < 64) {
        b1s[t] = b1[t]; b2s[t] = b2[t]; b3s[t] = b3[t];
        w1s[t] = W1[t]; w1s[64 + t] = W1[64 + t]; w1s[128 + t] = W1[128 + t];
    }
    __syncthreads();

    int wave = t >> 5, lane = t & 31;
    int tile = blockIdx.x * 8 + wave;
    if (tile >= numTiles) return;

    int m  = lane & 15;
    int hi = lane >> 4;          // 0: lanes 0-15, 1: lanes 16-31
    int e  = tile * 16 + m;
    bool valid = (e < E);
    int ec = valid ? e : (E - 1);

    int esrc = edge_index[ec];
    int edst = edge_index[E + ec];
    f2 xv = *(const f2*)(x + (long long)esrc * 2);
    float ea = edge_attr[ec];
    if (hi == 0) dsts[wave][m] = edst;

    // ---- layer 1 (K=3) straight into WMMA A-layout, 2 K-chunks of 32 ----
    v16bf A0 = {}, A1 = {};
#pragma unroll
    for (int j = 0; j < 16; ++j) {
        int kk = j + ((j >= 8) ? 8 : 0) + hi * 8;           // K within chunk
        {
            int k = kk;                                      // chunk 0
            float v = b1s[k] + xv.x * w1s[k] + xv.y * w1s[64 + k] + ea * w1s[128 + k];
            A0[j] = (__bf16)fmaxf(v, 0.0f);
        }
        {
            int k = 32 + kk;                                 // chunk 1
            float v = b1s[k] + xv.x * w1s[k] + xv.y * w1s[64 + k] + ea * w1s[128 + k];
            A1[j] = (__bf16)fmaxf(v, 0.0f);
        }
    }

    // ---- GEMM1: h2 = relu(h1 @ W2 + b2), stage bf16 into LDS ----
#pragma unroll
    for (int nt = 0; nt < 4; ++nt) {
        BFrag B0, B1;
        const __bf16* p0 = &w2s[(nt * 16 + m) * W23S + hi * 8];       // N row, K = 8hi..
        const __bf16* p1 = &w2s[(nt * 16 + m) * W23S + 32 + hi * 8];  // N row, K = 32+8hi..
        B0.q[0] = *(const vu4*)p0; B0.q[1] = *(const vu4*)(p0 + 16);
        B1.q[0] = *(const vu4*)p1; B1.q[1] = *(const vu4*)(p1 + 16);
        v8f acc = {};
        acc = __builtin_amdgcn_wmma_f32_16x16x32_bf16(false, A0, false, B0.v, (short)0, acc, false, false);
        acc = __builtin_amdgcn_wmma_f32_16x16x32_bf16(false, A1, false, B1.v, (short)0, acc, false, false);
        int n = nt * 16 + m;
        float bias = b2s[n];
#pragma unroll
        for (int r = 0; r < 8; ++r) {
            int mrow = r + hi * 8;
            h2s[wave][mrow * H2S + n] = (__bf16)fmaxf(acc[r] + bias, 0.0f);
        }
    }
    asm volatile("s_wait_dscnt 0" ::: "memory");

    // ---- reload h2 tile in A-layout ----
    AFrag a20, a21;
#pragma unroll
    for (int p = 0; p < 8; ++p) {
        int kk = 2 * p + ((p >= 4) ? 8 : 0) + hi * 8;
        a20.u[p] = *(const unsigned*)&h2s[wave][m * H2S + kk];
        a21.u[p] = *(const unsigned*)&h2s[wave][m * H2S + 32 + kk];
    }

    // ---- GEMM2: h3 = h2 @ W3 + b3, then atomic scatter-add ----
#pragma unroll
    for (int nt = 0; nt < 4; ++nt) {
        BFrag B0, B1;
        const __bf16* p0 = &w3s[(nt * 16 + m) * W23S + hi * 8];
        const __bf16* p1 = &w3s[(nt * 16 + m) * W23S + 32 + hi * 8];
        B0.q[0] = *(const vu4*)p0; B0.q[1] = *(const vu4*)(p0 + 16);
        B1.q[0] = *(const vu4*)p1; B1.q[1] = *(const vu4*)(p1 + 16);
        v8f acc = {};
        acc = __builtin_amdgcn_wmma_f32_16x16x32_bf16(false, a20.v, false, B0.v, (short)0, acc, false, false);
        acc = __builtin_amdgcn_wmma_f32_16x16x32_bf16(false, a21.v, false, B1.v, (short)0, acc, false, false);
        int n = nt * 16 + m;
        float bias = b3s[n];
#pragma unroll
        for (int r = 0; r < 8; ++r) {
            int mrow = r + hi * 8;
            if (tile * 16 + mrow < E) {
                int ed = dsts[wave][mrow];
                atomic_add_f32_asm(summed + (long long)ed * HID + n, acc[r] + bias);
            }
        }
    }
    if (hi == 0 && valid) atomic_add_f32_asm(cnt + edst, 1.0f);
}

// ---------------------------------------------------------------------------
// Node MLP: z=[x,agg,u]; z1=relu(z@W4+b4); out=z1@W5+b5.
// agg@W4-block via WMMA (N padded to 80), rank-3 rows + final layer via VALU.
// ---------------------------------------------------------------------------
__global__ void __launch_bounds__(256) node_kernel(
    const float* __restrict__ x, const int* __restrict__ batch,
    const float* __restrict__ u,
    const float* __restrict__ summed, const float* __restrict__ cnt,
    const __bf16* __restrict__ w4b,
    const float* __restrict__ W4, const float* __restrict__ b4,
    const float* __restrict__ W5, const float* __restrict__ b5,
    float* __restrict__ out, int N, int numTiles)
{
    __shared__ __bf16 w4s[64 * W4S];
    __shared__ float  w4r[3 * 80];     // W4 rows 0,1,66 (x0,x1,u), N-padded
    __shared__ float  b4s[80];
    __shared__ float  w5s[67 * 2];
    __shared__ float  b5s[2];
    __shared__ float  z1s[8][16 * Z1S];
    __shared__ float  xs[8][16 * 4];   // x0,x1,u,inv per row

    int t = threadIdx.x;
    for (int i = t; i < 64 * W4S; i += 256) w4s[i] = w4b[i];
    if (t < 80) {
        b4s[t]       = (t < 67) ? b4[t]           : 0.0f;
        w4r[t]       = (t < 67) ? W4[t]           : 0.0f;
        w4r[80 + t]  = (t < 67) ? W4[67 + t]      : 0.0f;
        w4r[160 + t] = (t < 67) ? W4[66 * 67 + t] : 0.0f;
    }
    if (t < 134) w5s[t] = W5[t];
    if (t < 2)   b5s[t] = b5[t];
    __syncthreads();

    int wave = t >> 5, lane = t & 31, m = lane & 15, hi = lane >> 4;
    int tile = blockIdx.x * 8 + wave;
    if (tile >= numTiles) return;

    int node = tile * 16 + m;
    int nc = (node < N) ? node : (N - 1);
    if (hi == 0) {
        float c = cnt[nc];
        xs[wave][m * 4 + 0] = x[(long long)nc * 2];
        xs[wave][m * 4 + 1] = x[(long long)nc * 2 + 1];
        xs[wave][m * 4 + 2] = u[batch[nc]];
        xs[wave][m * 4 + 3] = 1.0f / fmaxf(c, 1.0f);
    }
    asm volatile("s_wait_dscnt 0" ::: "memory");
    float inv = xs[wave][m * 4 + 3];

    // A fragments: agg row for this node, in WMMA A-layout
    const float* srow = summed + (long long)nc * HID;
    v16bf A0 = {}, A1 = {};
#pragma unroll
    for (int p = 0; p < 8; ++p) {
        int kk = 2 * p + ((p >= 4) ? 8 : 0) + hi * 8;
        f2 v0 = *(const f2*)(srow + kk);
        f2 v1 = *(const f2*)(srow + 32 + kk);
        A0[2 * p]     = (__bf16)(v0.x * inv);
        A0[2 * p + 1] = (__bf16)(v0.y * inv);
        A1[2 * p]     = (__bf16)(v1.x * inv);
        A1[2 * p + 1] = (__bf16)(v1.y * inv);
    }

#pragma unroll
    for (int nt = 0; nt < 5; ++nt) {
        BFrag B0, B1;
        const __bf16* p0 = &w4s[(nt * 16 + m) * 64 + hi * 8];
        const __bf16* p1 = &w4s[(nt * 16 + m) * 64 + 32 + hi * 8];
        B0.q[0] = *(const vu4*)p0; B0.q[1] = *(const vu4*)(p0 + 16);
        B1.q[0] = *(const vu4*)p1; B1.q[1] = *(const vu4*)(p1 + 16);
        v8f acc = {};
        acc = __builtin_amdgcn_wmma_f32_16x16x32_bf16(false, A0, false, B0.v, (short)0, acc, false, false);
        acc = __builtin_amdgcn_wmma_f32_16x16x32_bf16(false, A1, false, B1.v, (short)0, acc, false, false);
        int n = nt * 16 + m;
        if (n < 67) {
            float wa = w4r[n], wb = w4r[80 + n], wc = w4r[160 + n], bb = b4s[n];
#pragma unroll
            for (int r = 0; r < 8; ++r) {
                int mrow = r + hi * 8;
                const float* xr = &xs[wave][mrow * 4];
                float v = acc[r] + bb + xr[0] * wa + xr[1] * wb + xr[2] * wc;
                z1s[wave][mrow * Z1S + n] = fmaxf(v, 0.0f);
            }
        }
    }
    asm volatile("s_wait_dscnt 0" ::: "memory");

    if (hi == 0 && node < N) {
        const float* z = &z1s[wave][m * Z1S];
        float o0 = b5s[0], o1 = b5s[1];
#pragma unroll
        for (int k = 0; k < 67; ++k) {
            float zv = z[k];
            o0 += zv * w5s[2 * k];
            o1 += zv * w5s[2 * k + 1];
        }
        out[(long long)node * 2]     = o0;
        out[(long long)node * 2 + 1] = o1;
    }
}

// ---------------------------------------------------------------------------
extern "C" void kernel_launch(void* const* d_in, const int* in_sizes, int n_in,
                              void* d_out, int out_size, void* d_ws, size_t ws_size,
                              hipStream_t stream) {
    const float* x          = (const float*)d_in[0];
    const int*   edge_index = (const int*)  d_in[1];
    const float* edge_attr  = (const float*)d_in[2];
    const float* u          = (const float*)d_in[3];
    const int*   batch      = (const int*)  d_in[4];
    const float* W1 = (const float*)d_in[5];
    const float* b1 = (const float*)d_in[6];
    const float* W2 = (const float*)d_in[7];
    const float* b2 = (const float*)d_in[8];
    const float* W3 = (const float*)d_in[9];
    const float* b3 = (const float*)d_in[10];
    const float* W4 = (const float*)d_in[11];
    const float* b4 = (const float*)d_in[12];
    const float* W5 = (const float*)d_in[13];
    const float* b5 = (const float*)d_in[14];

    int N = in_sizes[4];   // batch has N_NODES entries
    int E = in_sizes[2];   // edge_attr has N_EDGES entries

    char* ws = (char*)d_ws;
    float*  summed = (float*)ws;
    float*  cntp   = (float*)(ws + (size_t)N * HID * 4);
    __bf16* w2b    = (__bf16*)(ws + (size_t)N * HID * 4 + (size_t)N * 4);
    __bf16* w3b    = w2b + 64 * W23S;
    __bf16* w4b    = w3b + 64 * W23S;

    long long ztot = (long long)N * HID;
    int prepBlocks = (int)((ztot + 255) / 256);
    prep_kernel<<<prepBlocks, 256, 0, stream>>>(W2, W3, W4, summed, cntp, w2b, w3b, w4b, N);

    int eTiles = (E + 15) / 16;
    int eBlocks = (eTiles + 7) / 8;
    edge_kernel<<<eBlocks, 256, 0, stream>>>(x, edge_index, edge_attr, W1, b1, b2, b3,
                                             w2b, w3b, summed, cntp, E, eTiles);

    int nTiles = (N + 15) / 16;
    int nBlocks = (nTiles + 7) / 8;
    node_kernel<<<nBlocks, 256, 0, stream>>>(x, batch, u, summed, cntp, w4b,
                                             W4, b4, W5, b5, (float*)d_out, N, nTiles);
}
